// ModifiedAttention_78331613544866
// MI455X (gfx1250) — compile-verified, matrix operand order checked by reading, not run on hardware
//
#include <hip/hip_runtime.h>
#include <hip/hip_bf16.h>

// ---------------------------------------------------------------------------
// ModifiedAttention for MI455X (gfx1250): bf16 WMMA, f32 accumulate,
// fused flash-attention with async double-buffered LDS staging of K/V.
// ---------------------------------------------------------------------------

#define DIMC   768
#define NHEAD  6
#define HD     128
#define BATCH  4
#define SEQ    2048
#define MROWS  (BATCH * SEQ)          // 8192
#define QKVO   (3 * DIMC)             // 2304
#define QSCALE 0.08838834764831845f   // 128^-0.5

// padded LDS row strides (elements) -> conflict-free ds_load_b128 fragments
#define KSTR 136                      // 32x128 K tile rows (272B stride)
#define VSTR 40                       // 128x32 V tile rows (80B stride)
#define PSTR 40                       // 16x32 P tile rows

typedef __attribute__((ext_vector_type(16))) __bf16 v16bf;
typedef __attribute__((ext_vector_type(8)))  __bf16 bf16x8;
typedef __attribute__((ext_vector_type(8)))  float  v8f;
typedef __attribute__((ext_vector_type(4)))  int    v4i;

#if __has_builtin(__builtin_amdgcn_global_load_async_to_lds_b128)
#define USE_ASYNC_LDS 1
#else
#define USE_ASYNC_LDS 0
#endif

union FragU { v16bf v; bf16x8 h[2]; };

// Load one 16x32 bf16 A/B fragment (ISA 7.12.2 layout).
// `p` points at (row for this lane) + k0 within a row-major source.
// lanes 0-15 take k {0..7,16..23}; lanes 16-31 take k {8..15,24..31}.
__device__ __forceinline__ v16bf load_frag(const __bf16* p, int lane) {
    p += (lane & 16) ? 8 : 0;
    FragU f;
    f.h[0] = *(const bf16x8*)p;
    f.h[1] = *(const bf16x8*)(p + 16);
    return f.v;
}

__device__ __forceinline__ v8f wmma_bf16(v16bf a, v16bf b, v8f c) {
    // (neg_a, A, neg_b, B, c_mod, C, reuse_a, reuse_b)
    return __builtin_amdgcn_wmma_f32_16x16x32_bf16(false, a, false, b,
                                                   (short)0, c, false, false);
}

// 16-byte global -> LDS copy: async (ASYNCcnt) when available.
__device__ __forceinline__ void copy16(const __bf16* g, __bf16* l) {
#if USE_ASYNC_LDS
    __builtin_amdgcn_global_load_async_to_lds_b128(
        (__attribute__((address_space(1))) v4i*)g,
        (__attribute__((address_space(3))) v4i*)l, 0, 0);
#else
    *(bf16x8*)l = *(const bf16x8*)g;
#endif
}

__device__ __forceinline__ void async_wait0() {
#if USE_ASYNC_LDS
#if __has_builtin(__builtin_amdgcn_s_wait_asynccnt)
    __builtin_amdgcn_s_wait_asynccnt(0);
#else
    asm volatile("s_wait_asynccnt 0x0" ::: "memory");
#endif
#endif
}

// ---------------------------------------------------------------------------
// Kernel 0: straight-through Gumbel-sigmoid router mask (forward == y_hard).
// ---------------------------------------------------------------------------
__global__ void mask_kernel(const float* __restrict__ latency,
                            const float* __restrict__ rw,
                            const float* __restrict__ rb,
                            const float* __restrict__ g1,
                            const float* __restrict__ g2,
                            float* __restrict__ maskC) {
    int c = blockIdx.x * blockDim.x + threadIdx.x;
    if (c < DIMC) {
        int h6 = c / HD;
        int j  = (h6 <= 3) ? 0 : (h6 - 3);   // mask6 = [m0,m0,m0,m0,m1,m2]
        float logit = rw[j] * latency[0] + rb[j] + g1[j] - g2[j];
        float y = 1.0f / (1.0f + __expf(-logit / 5.0f));
        maskC[c] = (y > 0.5f) ? 1.0f : 0.0f;
    }
}

// ---------------------------------------------------------------------------
// f32 -> bf16 conversion
// ---------------------------------------------------------------------------
__global__ void cvt_kernel(const float* __restrict__ in,
                           __bf16* __restrict__ out, int n) {
    int i = blockIdx.x * blockDim.x + threadIdx.x;
    if (i < n) out[i] = (__bf16)in[i];
}

// ---------------------------------------------------------------------------
// GEMM: out = A(bf16, MxK row-major) @ W^T(bf16, OxK row-major) + bias.
// Block = 256 threads = 8 waves, tile 128(M) x 64(N); wave tile 32x32.
// mode 0: QKV epilogue -> scatter Q(scaled)/K [B,H,N,HD], V transposed
//         [B,H,HD,N], all bf16.
// mode 1: proj epilogue -> f32 d_out + bias.
// ---------------------------------------------------------------------------
__global__ __launch_bounds__(256) void gemm_wmma(
    const __bf16* __restrict__ A, const __bf16* __restrict__ W,
    const float* __restrict__ bias, int K, int mode,
    __bf16* __restrict__ qb, __bf16* __restrict__ kb,
    __bf16* __restrict__ vt, float* __restrict__ out) {

    const int lane = threadIdx.x & 31;
    const int w    = threadIdx.x >> 5;
    const int wm   = w & 3;           // 4 waves along M
    const int wn   = w >> 2;          // 2 waves along N
    const int m0   = blockIdx.x * 128 + wm * 32;
    const int n0   = blockIdx.y * 64 + wn * 32;
    const int lrow = lane & 15;

    v8f acc[2][2] = {};
    for (int k0 = 0; k0 < K; k0 += 32) {
        v16bf a[2], b[2];
        a[0] = load_frag(A + (size_t)(m0 + lrow)      * K + k0, lane);
        a[1] = load_frag(A + (size_t)(m0 + 16 + lrow) * K + k0, lane);
        b[0] = load_frag(W + (size_t)(n0 + lrow)      * K + k0, lane);
        b[1] = load_frag(W + (size_t)(n0 + 16 + lrow) * K + k0, lane);
#pragma unroll
        for (int i = 0; i < 2; ++i)
#pragma unroll
            for (int j = 0; j < 2; ++j)
                acc[i][j] = wmma_bf16(a[i], b[j], acc[i][j]);
    }

    const int halfoff = (lane & 16) ? 8 : 0;
#pragma unroll
    for (int i = 0; i < 2; ++i)
#pragma unroll
        for (int j = 0; j < 2; ++j) {
            const int cbase = n0 + 16 * j + lrow;
#pragma unroll
            for (int p = 0; p < 8; ++p) {
                const int r = m0 + 16 * i + p + halfoff;
                float val = acc[i][j][p] + bias[cbase];
                if (mode == 0) {
                    int which = cbase / DIMC;
                    int rem   = cbase - which * DIMC;
                    int h     = rem / HD;
                    int d     = rem & (HD - 1);
                    int bi    = r >> 11;          // / SEQ
                    int ns    = r & (SEQ - 1);
                    size_t bh = (size_t)(bi * NHEAD + h);
                    if (which == 0)
                        qb[(bh * SEQ + ns) * HD + d] = (__bf16)(val * QSCALE);
                    else if (which == 1)
                        kb[(bh * SEQ + ns) * HD + d] = (__bf16)val;
                    else
                        vt[(bh * HD + d) * SEQ + ns] = (__bf16)val;
                } else {
                    out[(size_t)r * DIMC + cbase] = val;
                }
            }
        }
}

// ---------------------------------------------------------------------------
// Flash attention. One block = 8 waves = 128 query rows; K/V tiles for each
// 32-key step are staged once per block into double-buffered LDS (async
// load-to-LDS, ASYNCcnt), then each wave runs 8 QK WMMAs, online softmax,
// P->A-layout via its private LDS tile, and 8 PV WMMAs.
// ---------------------------------------------------------------------------
__global__ __launch_bounds__(256) void attn_wmma(
    const __bf16* __restrict__ qb, const __bf16* __restrict__ kb,
    const __bf16* __restrict__ vt, const float* __restrict__ maskC,
    __bf16* __restrict__ ao) {

    __shared__ __align__(16) __bf16 sK[2][32 * KSTR];   // 2 x 8.5 KB
    __shared__ __align__(16) __bf16 sV[2][128 * VSTR];  // 2 x 10  KB
    __shared__ __align__(16) __bf16 pls[8][16 * PSTR];  // 10 KB (per-wave P)

    const int tid  = threadIdx.x;
    const int lane = tid & 31;
    const int w    = tid >> 5;
    const int bh   = blockIdx.x >> 4;                  // 16 q-chunks per (b,h)
    const int q0   = (blockIdx.x & 15) * 128 + w * 16;
    const int lrow = lane & 15;
    const int halfoff = (lane & 16) ? 8 : 0;

    const __bf16* qbh = qb + (size_t)bh * SEQ * HD;
    const __bf16* kbh = kb + (size_t)bh * SEQ * HD;
    const __bf16* vbh = vt + (size_t)bh * HD * SEQ;
    __bf16* pw = &pls[w][0];

    // stage one 32-key step (K: 32x128 row-major, V: 128x32 from transposed V)
    auto stage = [&](int kt, int buf) {
#pragma unroll
        for (int r = 0; r < 2; ++r) {                  // K tile: 512 chunks
            int c = tid + 256 * r;
            int row = c >> 4, cc = c & 15;
            copy16(kbh + (size_t)(kt + row) * HD + cc * 8,
                   &sK[buf][row * KSTR + cc * 8]);
        }
#pragma unroll
        for (int r = 0; r < 2; ++r) {                  // V tile: 512 chunks
            int c = tid + 256 * r;
            int row = c >> 2, cc = c & 3;
            copy16(vbh + (size_t)row * SEQ + kt + cc * 8,
                   &sV[buf][row * VSTR + cc * 8]);
        }
    };

    v16bf aq[4];
#pragma unroll
    for (int t = 0; t < 4; ++t)
        aq[t] = load_frag(qbh + (size_t)(q0 + lrow) * HD + 32 * t, lane);

    v8f accO[8] = {};
    float rm[8], rs[8];
#pragma unroll
    for (int p = 0; p < 8; ++p) { rm[p] = -3.0e38f; rs[p] = 0.0f; }

    stage(0, 0);
    async_wait0();
    __syncthreads();

    for (int kt = 0; kt < SEQ; kt += 32) {
        const int buf = (kt >> 5) & 1;
        if (kt + 32 < SEQ) stage(kt + 32, buf ^ 1);    // prefetch next step

        const __bf16* sKb = &sK[buf][0];
        const __bf16* sVb = &sV[buf][0];

        v8f s0 = {}, s1 = {};
#pragma unroll
        for (int t = 0; t < 4; ++t) {
            v16bf b0 = load_frag(sKb + (lrow)      * KSTR + 32 * t, lane);
            v16bf b1 = load_frag(sKb + (16 + lrow) * KSTR + 32 * t, lane);
            s0 = wmma_bf16(aq[t], b0, s0);
            s1 = wmma_bf16(aq[t], b1, s1);
        }

        float alpha[8];
#pragma unroll
        for (int p = 0; p < 8; ++p) {
            float mx = fmaxf(s0[p], s1[p]);
            mx = fmaxf(mx, __shfl_xor(mx, 1, 32));
            mx = fmaxf(mx, __shfl_xor(mx, 2, 32));
            mx = fmaxf(mx, __shfl_xor(mx, 4, 32));
            mx = fmaxf(mx, __shfl_xor(mx, 8, 32));
            float mn = fmaxf(rm[p], mx);
            float al = __expf(rm[p] - mn);
            float p0 = __expf(s0[p] - mn);
            float p1 = __expf(s1[p] - mn);
            float sm = p0 + p1;
            sm += __shfl_xor(sm, 1, 32);
            sm += __shfl_xor(sm, 2, 32);
            sm += __shfl_xor(sm, 4, 32);
            sm += __shfl_xor(sm, 8, 32);
            rs[p] = rs[p] * al + sm;
            rm[p] = mn;
            alpha[p] = al;
            const int row = p + halfoff;
            pw[row * PSTR + lrow]      = (__bf16)p0;
            pw[row * PSTR + 16 + lrow] = (__bf16)p1;
        }
#pragma unroll
        for (int j = 0; j < 8; ++j)
#pragma unroll
            for (int p = 0; p < 8; ++p)
                accO[j][p] *= alpha[p];

        // P (C-layout f32) -> A-layout bf16 via per-wave LDS tile; LDS ops
        // are in-order within a wave (DScnt).
        v16bf ap = load_frag(pw + lrow * PSTR, lane);
#pragma unroll
        for (int j = 0; j < 8; ++j) {
            v16bf bv = load_frag(sVb + (16 * j + lrow) * VSTR, lane);
            accO[j] = wmma_bf16(ap, bv, accO[j]);
        }

        async_wait0();       // our slice of next tiles landed
        __syncthreads();     // everyone's slice landed; safe to swap buffers
    }

    const int h = bh % NHEAD;
    const int b = bh / NHEAD;
#pragma unroll
    for (int j = 0; j < 8; ++j) {
        const int d = 16 * j + lrow;
        const float msk = maskC[h * HD + d];
#pragma unroll
        for (int p = 0; p < 8; ++p) {
            const int row = q0 + p + halfoff;
            float val = accO[j][p] / rs[p] * msk;
            ao[((size_t)(b * SEQ + row)) * DIMC + h * HD + d] = (__bf16)val;
        }
    }
}

// ---------------------------------------------------------------------------
extern "C" void kernel_launch(void* const* d_in, const int* in_sizes, int n_in,
                              void* d_out, int out_size, void* d_ws, size_t ws_size,
                              hipStream_t stream) {
    const float* x        = (const float*)d_in[0];
    const float* latency  = (const float*)d_in[1];
    const float* qkv_w    = (const float*)d_in[2];
    const float* qkv_b    = (const float*)d_in[3];
    const float* proj_w   = (const float*)d_in[4];
    const float* proj_b   = (const float*)d_in[5];
    const float* router_w = (const float*)d_in[6];
    const float* router_b = (const float*)d_in[7];
    const float* g1       = (const float*)d_in[8];
    const float* g2       = (const float*)d_in[9];
    float* out = (float*)d_out;

    const size_t XN = (size_t)MROWS * DIMC;        // 6291456
    const size_t QN = (size_t)BATCH * NHEAD * SEQ * HD;

    char* ws = (char*)d_ws;
    size_t off = 0;
    auto take = [&](size_t bytes) -> void* {
        void* p = ws + off;
        off = (off + bytes + 255) & ~(size_t)255;
        return p;
    };
    float*  maskC = (float*) take(DIMC * sizeof(float));
    __bf16* xb    = (__bf16*)take(XN * 2);
    __bf16* wqkv  = (__bf16*)take((size_t)QKVO * DIMC * 2);
    __bf16* wproj = (__bf16*)take((size_t)DIMC * DIMC * 2);
    __bf16* q_b   = (__bf16*)take(QN * 2);
    __bf16* k_b   = (__bf16*)take(QN * 2);
    __bf16* v_t   = (__bf16*)take(QN * 2);
    __bf16* ao    = (__bf16*)take(XN * 2);

    mask_kernel<<<3, 256, 0, stream>>>(latency, router_w, router_b, g1, g2, maskC);
    cvt_kernel<<<(int)((XN + 255) / 256), 256, 0, stream>>>(x, xb, (int)XN);
    cvt_kernel<<<(QKVO * DIMC + 255) / 256, 256, 0, stream>>>(qkv_w, wqkv, QKVO * DIMC);
    cvt_kernel<<<(DIMC * DIMC + 255) / 256, 256, 0, stream>>>(proj_w, wproj, DIMC * DIMC);

    // QKV: M=8192, N=2304, K=768
    gemm_wmma<<<dim3(MROWS / 128, QKVO / 64), 256, 0, stream>>>(
        xb, wqkv, qkv_b, DIMC, 0, q_b, k_b, v_t, nullptr);

    // Attention: 24 (b,h) pairs x 16 q-chunks of 128 rows
    attn_wmma<<<BATCH * NHEAD * (SEQ / 128), 256, 0, stream>>>(
        q_b, k_b, v_t, maskC, ao);

    // Projection: M=8192, N=768, K=768 -> f32 out + bias
    gemm_wmma<<<dim3(MROWS / 128, DIMC / 64), 256, 0, stream>>>(
        ao, wproj, proj_b, DIMC, 1, nullptr, nullptr, nullptr, out);
}
